// SpatialMultiheadAttention_16329465660112
// MI455X (gfx1250) — matrix operand order repaired and checked by reading, not compiled, not run on hardware
//
#include <hip/hip_runtime.h>

typedef __attribute__((ext_vector_type(16))) __bf16        v16bf;
typedef __attribute__((ext_vector_type(8)))  float         v8f;
typedef __attribute__((ext_vector_type(4))) unsigned int   u32x4;
typedef __attribute__((ext_vector_type(4)))  float         f32x4;
typedef int v4i __attribute__((vector_size(16)));

constexpr int Bc = 2, Sc = 2048, Dc = 1024, Hc = 16, Kc = 64;
constexpr float NEG_BIG = -3.402823466e38f;   // finfo(float32).min

// ---------------- CDNA5 async global->LDS path (guarded) ----------------

#if defined(__has_builtin)
#if __has_builtin(__builtin_amdgcn_global_load_async_to_lds_b128) && \
    __has_builtin(__builtin_amdgcn_s_wait_asynccnt)
#define HAVE_ASYNC 1
#endif
#endif
#ifndef HAVE_ASYNC
#define HAVE_ASYNC 0
#endif

#define GAS __attribute__((address_space(1)))
#define LAS __attribute__((address_space(3)))

// copy 16 bytes global -> LDS (async DMA when available, sync fallback)
__device__ __forceinline__ void async_copy16(const void* g, void* l) {
#if HAVE_ASYNC
  __builtin_amdgcn_global_load_async_to_lds_b128((GAS v4i*)(void*)g,
                                                 (LAS v4i*)l,
                                                 /*offset=*/0, /*cpol=*/0);
#else
  *(u32x4*)l = *(const u32x4*)g;
#endif
}

template <int N>
__device__ __forceinline__ void wait_async() {
#if HAVE_ASYNC
  __builtin_amdgcn_s_wait_asynccnt(N);
#endif
}

// ---------------- WMMA helpers (gfx1250 wave32 layouts) ----------------

__device__ __forceinline__ v8f wmma_bf16(v16bf a, v16bf b, v8f c) {
  return __builtin_amdgcn_wmma_f32_16x16x32_bf16(
      false, a, false, b, (short)0, c, false, false);
}

// A 16x32 bf16 fragment from row-major src.
__device__ __forceinline__ v16bf load_a_bf16(const __bf16* __restrict__ base,
                                             int ld, int row0, int k0, int lane) {
  int r  = lane & 15;
  int kb = (lane >> 4) * 8;
  const __bf16* p = base + (size_t)(row0 + r) * ld + k0 + kb;
  union { u32x4 u[2]; v16bf v; } t;
  t.u[0] = *(const u32x4*)(p);        // K = kb+0 .. kb+7
  t.u[1] = *(const u32x4*)(p + 16);   // K = kb+16 .. kb+23
  return t.v;
}

// A fragment from fp32 source, converted to bf16 in-register (for P·V).
__device__ __forceinline__ v16bf load_a_f32(const float* __restrict__ base,
                                            int ld, int row0, int k0, int lane) {
  int r  = lane & 15;
  int kb = (lane >> 4) * 8;
  const float* p = base + (size_t)(row0 + r) * ld + k0 + kb;
  f32x4 f0 = *(const f32x4*)(p);
  f32x4 f1 = *(const f32x4*)(p + 4);
  f32x4 f2 = *(const f32x4*)(p + 16);
  f32x4 f3 = *(const f32x4*)(p + 20);
  v16bf v;
  v[0]=(__bf16)f0.x; v[1]=(__bf16)f0.y; v[2]=(__bf16)f0.z; v[3]=(__bf16)f0.w;
  v[4]=(__bf16)f1.x; v[5]=(__bf16)f1.y; v[6]=(__bf16)f1.z; v[7]=(__bf16)f1.w;
  v[8]=(__bf16)f2.x;  v[9]=(__bf16)f2.y;  v[10]=(__bf16)f2.z; v[11]=(__bf16)f2.w;
  v[12]=(__bf16)f3.x; v[13]=(__bf16)f3.y; v[14]=(__bf16)f3.z; v[15]=(__bf16)f3.w;
  return v;
}

// B 32x16 fragment from LDS where element (k, n) = lds[(n0+n)*stride + k0 + k].
// stride padded (40 or 72 elems) so 16 lanes hit 16 distinct banks.
__device__ __forceinline__ v16bf load_b_lds(const __bf16* lds, int strideElems,
                                            int n0, int k0, int lane) {
  int n  = lane & 15;
  int kb = (lane >> 4) * 8;
  const __bf16* p = lds + (n0 + n) * strideElems + k0 + kb;
  union { u32x4 u[2]; v16bf v; } t;
  t.u[0] = *(const u32x4*)(p);
  t.u[1] = *(const u32x4*)(p + 16);
  return t.v;
}

// Stage a 64-row x 32-elem bf16 block into LDS (row stride 40 elems = 80B).
// 64 rows * 4 x 16B parts = 256 tasks = one per thread.
__device__ __forceinline__ void stage_b_chunk(const __bf16* __restrict__ g, int ld,
                                              int n0, int k0, __bf16* lds, int tid) {
  int row  = tid >> 2;   // 0..63
  int part = tid & 3;    // 16B each
  async_copy16(g + (size_t)(n0 + row) * ld + k0 + part * 8,
               lds + row * 40 + part * 8);
}

// ---------------- precision-cast / transpose kernels ----------------

__global__ void k_cast(const float* __restrict__ src, __bf16* __restrict__ dst, int n) {
  int i = blockIdx.x * blockDim.x + threadIdx.x;
  if (i < n) dst[i] = (__bf16)src[i];
}

// W [HH, DD, KK] row-major  ->  Wt [HH, KK, DD]  (bf16)
__global__ void k_transpose_w(const float* __restrict__ W, __bf16* __restrict__ Wt,
                              int HH, int DD, int KK) {
  int i = blockIdx.x * blockDim.x + threadIdx.x;
  int total = HH * DD * KK;
  if (i >= total) return;
  int h = i / (DD * KK);
  int rem = i % (DD * KK);
  int d = rem / KK;
  int k = rem % KK;
  Wt[(size_t)h * KK * DD + (size_t)k * DD + d] = (__bf16)W[i];
}

// ---------------- QKV projection ----------------
// grid (S/128, H, 3*B), block 256 (8 waves). Weight chunk (64xK x 32d) staged
// in LDS, double-buffered with async copies; A fragments pipelined one ahead.

__global__ void k_qkv(const __bf16* __restrict__ xb,
                      const __bf16* __restrict__ wqt, const __bf16* __restrict__ wkt,
                      const __bf16* __restrict__ wvt,
                      const float* __restrict__ bq, const float* __restrict__ bk,
                      const float* __restrict__ bv,
                      __bf16* __restrict__ qo, __bf16* __restrict__ ko,
                      __bf16* __restrict__ vto) {
  __shared__ __bf16 sB[2][64 * 40];
  int tid = threadIdx.x;
  int lane = tid & 31, wave = tid >> 5;
  int h   = blockIdx.y;
  int sel = blockIdx.z / Bc;
  int b   = blockIdx.z % Bc;
  const __bf16* wt   = (sel == 0) ? wqt : (sel == 1) ? wkt : wvt;
  const float*  bias = (sel == 0) ? bq  : (sel == 1) ? bk  : bv;

  int s0 = (blockIdx.x * 8 + wave) * 16;
  const __bf16* A  = xb + (size_t)b * Sc * Dc;          // [S, D]
  const __bf16* Bm = wt + (size_t)h * Kc * Dc;          // Wt[h] : [K(=64), D]

  stage_b_chunk(Bm, Dc, 0, 0, sB[0], tid);
  v16bf a_cur = load_a_bf16(A, Dc, s0, 0, lane);

  v8f acc[4] = {};
  const int NCH = Dc / 32;
  for (int c = 0; c < NCH; ++c) {
    bool more = (c + 1) < NCH;
    if (more) stage_b_chunk(Bm, Dc, 0, (c + 1) * 32, sB[(c + 1) & 1], tid);
    v16bf a_next = more ? load_a_bf16(A, Dc, s0, (c + 1) * 32, lane) : a_cur;
    if (more) wait_async<1>(); else wait_async<0>();
    __syncthreads();
    const __bf16* buf = sB[c & 1];
#pragma unroll
    for (int nt = 0; nt < 4; ++nt)
      acc[nt] = wmma_bf16(a_cur, load_b_lds(buf, 40, nt * 16, 0, lane), acc[nt]);
    __syncthreads();
    a_cur = a_next;
  }

  int n = lane & 15, hi = lane >> 4;
#pragma unroll
  for (int nt = 0; nt < 4; ++nt) {
    int kcol = nt * 16 + n;
    float bval = bias[h * Kc + kcol];
    if (sel == 2) {
      // V^T [B,H,K,S]: 8 consecutive s per lane -> one 16B store
      union { u32x4 u; __bf16 e[8]; } pk;
#pragma unroll
      for (int i = 0; i < 8; ++i) pk.e[i] = (__bf16)(acc[nt][i] + bval);
      *(u32x4*)(vto + ((size_t)(b * Hc + h) * Kc + kcol) * Sc + s0 + 8 * hi) = pk.u;
    } else {
      __bf16* o = sel ? ko : qo;
#pragma unroll
      for (int i = 0; i < 8; ++i) {
        int s = s0 + i + 8 * hi;
        o[((size_t)(b * Hc + h) * Sc + s) * Kc + kcol] = (__bf16)(acc[nt][i] + bval);
      }
    }
  }
}

// ---------------- scores = scale * Q K^T, masked -> d_out (fp32) ----------
// grid (S/128, S/128, B*H), block 256. K-block (128 t x 64 k) staged in LDS
// once per block via async copies; each wave then does 16 WMMAs from LDS.

__global__ void k_scores(const __bf16* __restrict__ qb, const __bf16* __restrict__ kb,
                         const int* __restrict__ mask, float* __restrict__ scores) {
  __shared__ __bf16 sK[128 * 72];
  int tid = threadIdx.x;
  int lane = tid & 31, wave = tid >> 5;
  int b = blockIdx.z / Hc, h = blockIdx.z % Hc;
  int s0  = blockIdx.y * 128 + wave * 16;
  int t0b = blockIdx.x * 128;

  const __bf16* Q  = qb + (size_t)(b * Hc + h) * Sc * Kc;   // [S, 64]
  const __bf16* Km = kb + (size_t)(b * Hc + h) * Sc * Kc;   // [S, 64]

  // stage 128x64 K block: 128 rows * 8 parts = 1024 tasks, 4 per thread
#pragma unroll
  for (int r = 0; r < 4; ++r) {
    int task = tid + (r << 8);
    int row  = task >> 3;
    int part = task & 7;
    async_copy16(Km + (size_t)(t0b + row) * Kc + part * 8,
                 sK + row * 72 + part * 8);
  }

  v16bf a0 = load_a_bf16(Q, Kc, s0, 0,  lane);
  v16bf a1 = load_a_bf16(Q, Kc, s0, 32, lane);

  // prefetch the mask block this wave will read in the epilogue
  int n = lane & 15, hi = lane >> 4;
  {
    const int* mrow = mask + ((size_t)b * Sc + s0 + (lane & 15)) * Sc + t0b + hi * 64;
    __builtin_prefetch(mrow, 0, 3);
    __builtin_prefetch(mrow + 32, 0, 3);
  }

  wait_async<0>();
  __syncthreads();

  v8f acc[8] = {};
#pragma unroll
  for (int nt = 0; nt < 8; ++nt) {
    acc[nt] = wmma_bf16(a0, load_b_lds(sK, 72, nt * 16, 0,  lane), acc[nt]);
    acc[nt] = wmma_bf16(a1, load_b_lds(sK, 72, nt * 16, 32, lane), acc[nt]);
  }

  const float scale = 0.125f;  // 1/sqrt(64)
  float* srow = scores + (size_t)(b * Hc + h) * Sc * Sc;
#pragma unroll
  for (int nt = 0; nt < 8; ++nt) {
    int t = t0b + nt * 16 + n;
#pragma unroll
    for (int i = 0; i < 8; ++i) {
      int s = s0 + i + 8 * hi;
      int m = mask[((size_t)b * Sc + s) * Sc + t];
      float v = m ? acc[nt][i] * scale : NEG_BIG;
      srow[(size_t)s * Sc + t] = v;
    }
  }
}

// ---------------- in-place row softmax over S=2048, one block per row -----

__global__ void k_softmax(float* __restrict__ sc) {
  __shared__ float red[256];
  float* p = sc + (size_t)blockIdx.x * Sc;
  int tid = threadIdx.x;

  f32x4 a = *(const f32x4*)(p + tid * 8);
  f32x4 c = *(const f32x4*)(p + tid * 8 + 4);
  float m = fmaxf(fmaxf(fmaxf(a.x, a.y), fmaxf(a.z, a.w)),
                  fmaxf(fmaxf(c.x, c.y), fmaxf(c.z, c.w)));
  red[tid] = m; __syncthreads();
  for (int s = 128; s > 0; s >>= 1) {
    if (tid < s) red[tid] = fmaxf(red[tid], red[tid + s]);
    __syncthreads();
  }
  float rmax = red[0]; __syncthreads();

  float e[8];
  e[0] = expf(a.x - rmax); e[1] = expf(a.y - rmax);
  e[2] = expf(a.z - rmax); e[3] = expf(a.w - rmax);
  e[4] = expf(c.x - rmax); e[5] = expf(c.y - rmax);
  e[6] = expf(c.z - rmax); e[7] = expf(c.w - rmax);
  float sum = e[0]+e[1]+e[2]+e[3]+e[4]+e[5]+e[6]+e[7];
  red[tid] = sum; __syncthreads();
  for (int s = 128; s > 0; s >>= 1) {
    if (tid < s) red[tid] += red[tid + s];
    __syncthreads();
  }
  float inv = 1.0f / red[0];

  a.x = e[0]*inv; a.y = e[1]*inv; a.z = e[2]*inv; a.w = e[3]*inv;
  c.x = e[4]*inv; c.y = e[5]*inv; c.z = e[6]*inv; c.w = e[7]*inv;
  *(f32x4*)(p + tid * 8)     = a;
  *(f32x4*)(p + tid * 8 + 4) = c;
}

// ---------------- heads = P (SxS fp32) * V  -> concat bf16 [B,S,H*K] ------
// grid (S/128, H, B). V^T chunk (64k x 32t) staged in LDS, double-buffered.

__global__ void k_pv(const float* __restrict__ scores, const __bf16* __restrict__ vtb,
                     __bf16* __restrict__ concat) {
  __shared__ __bf16 sV[2][64 * 40];
  int tid = threadIdx.x;
  int lane = tid & 31, wave = tid >> 5;
  int b = blockIdx.z, h = blockIdx.y;
  int s0 = (blockIdx.x * 8 + wave) * 16;

  const float*  P  = scores + (size_t)(b * Hc + h) * Sc * Sc;   // [S, S]
  const __bf16* Vt = vtb    + (size_t)(b * Hc + h) * Kc * Sc;   // [64, S]

  stage_b_chunk(Vt, Sc, 0, 0, sV[0], tid);
  v16bf a_cur = load_a_f32(P, Sc, s0, 0, lane);

  v8f acc[4] = {};
  const int NCH = Sc / 32;
  for (int c = 0; c < NCH; ++c) {
    bool more = (c + 1) < NCH;
    if (more) stage_b_chunk(Vt, Sc, 0, (c + 1) * 32, sV[(c + 1) & 1], tid);
    v16bf a_next = more ? load_a_f32(P, Sc, s0, (c + 1) * 32, lane) : a_cur;
    if (more) wait_async<1>(); else wait_async<0>();
    __syncthreads();
    const __bf16* buf = sV[c & 1];
#pragma unroll
    for (int nt = 0; nt < 4; ++nt)
      acc[nt] = wmma_bf16(a_cur, load_b_lds(buf, 40, nt * 16, 0, lane), acc[nt]);
    __syncthreads();
    a_cur = a_next;
  }

  int n = lane & 15, hi = lane >> 4;
#pragma unroll
  for (int nt = 0; nt < 4; ++nt) {
    int k = nt * 16 + n;
#pragma unroll
    for (int i = 0; i < 8; ++i) {
      int s = s0 + i + 8 * hi;
      concat[((size_t)b * Sc + s) * (Hc * Kc) + h * Kc + k] = (__bf16)acc[nt][i];
    }
  }
}

// ---------------- out = concat (4096x1024) * W_proj + b_proj (fp32) -------
// grid (D/64, (B*S)/128). W_proj^T chunk staged in LDS, double-buffered.

__global__ void k_proj(const __bf16* __restrict__ concat, const __bf16* __restrict__ wpt,
                       const float* __restrict__ bproj, float* __restrict__ out) {
  __shared__ __bf16 sW[2][64 * 40];
  int tid = threadIdx.x;
  int lane = tid & 31, wave = tid >> 5;
  int r0 = (blockIdx.y * 8 + wave) * 16;   // row in [0, B*S)
  int n0 = blockIdx.x * 64;                // output feature base

  stage_b_chunk(wpt, Hc * Kc, n0, 0, sW[0], tid);
  v16bf a_cur = load_a_bf16(concat, Hc * Kc, r0, 0, lane);

  v8f acc[4] = {};
  const int NCH = (Hc * Kc) / 32;
  for (int c = 0; c < NCH; ++c) {
    bool more = (c + 1) < NCH;
    if (more) stage_b_chunk(wpt, Hc * Kc, n0, (c + 1) * 32, sW[(c + 1) & 1], tid);
    v16bf a_next = more ? load_a_bf16(concat, Hc * Kc, r0, (c + 1) * 32, lane) : a_cur;
    if (more) wait_async<1>(); else wait_async<0>();
    __syncthreads();
    const __bf16* buf = sW[c & 1];
#pragma unroll
    for (int nt = 0; nt < 4; ++nt)
      acc[nt] = wmma_bf16(a_cur, load_b_lds(buf, 40, nt * 16, 0, lane), acc[nt]);
    __syncthreads();
    a_cur = a_next;
  }

  int n = lane & 15, hi = lane >> 4;
#pragma unroll
  for (int nt = 0; nt < 4; ++nt) {
    int col = n0 + nt * 16 + n;
    float bval = bproj[col];
#pragma unroll
    for (int i = 0; i < 8; ++i) {
      int r = r0 + i + 8 * hi;
      out[(size_t)r * Dc + col] = acc[nt][i] + bval;
    }
  }
}

// ---------------- host-side orchestration ----------------

extern "C" void kernel_launch(void* const* d_in, const int* in_sizes, int n_in,
                              void* d_out, int out_size, void* d_ws, size_t ws_size,
                              hipStream_t stream) {
  const float* x      = (const float*)d_in[0];
  const float* W_Q    = (const float*)d_in[1];
  const float* b_Q    = (const float*)d_in[2];
  const float* W_K    = (const float*)d_in[3];
  const float* b_K    = (const float*)d_in[4];
  const float* W_V    = (const float*)d_in[5];
  const float* b_V    = (const float*)d_in[6];
  const float* W_proj = (const float*)d_in[7];
  const float* b_proj = (const float*)d_in[8];
  const int*   mask   = (const int*)d_in[9];

  float* attn_out = (float*)d_out;                                   // [B,S,D]
  float* scores   = attn_out + (size_t)Bc * Sc * Dc;                 // [B,H,S,S]

  // workspace carve-up (bf16 elements)
  __bf16* p = (__bf16*)d_ws;
  __bf16* xb   = p; p += (size_t)Bc * Sc * Dc;          // x as bf16
  __bf16* wqt  = p; p += (size_t)Hc * Kc * Dc;          // W_Q^T [H,K,D]
  __bf16* wkt  = p; p += (size_t)Hc * Kc * Dc;
  __bf16* wvt  = p; p += (size_t)Hc * Kc * Dc;
  __bf16* wpt  = p; p += (size_t)Dc * Hc * Kc;          // W_proj^T [D, H*K]
  __bf16* qb   = p; p += (size_t)Bc * Hc * Sc * Kc;     // Q  [B,H,S,K]
  __bf16* kb   = p; p += (size_t)Bc * Hc * Sc * Kc;     // K  [B,H,S,K]
  __bf16* vtb  = p; p += (size_t)Bc * Hc * Kc * Sc;     // V^T[B,H,K,S]
  __bf16* conc = p; p += (size_t)Bc * Sc * Hc * Kc;     // concat [B,S,H*K]

  const int T = 256;
  int nx = Bc * Sc * Dc;
  k_cast<<<(nx + T - 1) / T, T, 0, stream>>>(x, xb, nx);

  int nw = Hc * Dc * Kc;
  k_transpose_w<<<(nw + T - 1) / T, T, 0, stream>>>(W_Q, wqt, Hc, Dc, Kc);
  k_transpose_w<<<(nw + T - 1) / T, T, 0, stream>>>(W_K, wkt, Hc, Dc, Kc);
  k_transpose_w<<<(nw + T - 1) / T, T, 0, stream>>>(W_V, wvt, Hc, Dc, Kc);
  int np = Hc * Kc * Dc;
  k_transpose_w<<<(np + T - 1) / T, T, 0, stream>>>(W_proj, wpt, 1, Hc * Kc, Dc);

  k_qkv<<<dim3(Sc / 128, Hc, 3 * Bc), T, 0, stream>>>(
      xb, wqt, wkt, wvt, b_Q, b_K, b_V, qb, kb, vtb);

  k_scores<<<dim3(Sc / 128, Sc / 128, Bc * Hc), T, 0, stream>>>(qb, kb, mask, scores);

  k_softmax<<<Bc * Hc * Sc, T, 0, stream>>>(scores);

  k_pv<<<dim3(Sc / 128, Hc, Bc), T, 0, stream>>>(scores, vtb, conc);

  k_proj<<<dim3(Dc / 64, (Bc * Sc) / 128), T, 0, stream>>>(conc, wpt, b_proj, attn_out);
}
